// SwitchTransformerLayer_11330123727003
// MI455X (gfx1250) — matrix operand order repaired and checked by reading, not compiled, not run on hardware
//
#include <hip/hip_runtime.h>
#include <math.h>
#include <stdint.h>

// Problem sizes (fixed by the reference)
#define BB 4
#define SS 2048
#define NN (BB * SS)   // 8192 tokens
#define DD 1024
#define FF 4096
#define EE 8

typedef __attribute__((ext_vector_type(16))) __bf16 v16bf;
typedef __attribute__((ext_vector_type(8)))  __bf16 v8bf;
typedef __attribute__((ext_vector_type(4)))  __bf16 v4bf;
typedef __attribute__((ext_vector_type(2)))  __bf16 v2bf;
typedef __attribute__((ext_vector_type(8)))  float  v8f;

// ---------------------------------------------------------------------------
// CDNA5 async copy: 16B global -> LDS via the async-DMA path (ASYNCcnt).
// ---------------------------------------------------------------------------
__device__ __forceinline__ void async_copy_b128(const void* g, void* l) {
  const uint64_t ga = (uint64_t)(uintptr_t)g;
  const uint32_t la = (uint32_t)(uintptr_t)l;   // generic addr low 32b == LDS offset
  asm volatile("global_load_async_to_lds_b128 %0, %1, off"
               :: "v"(la), "v"(ga) : "memory");
}

__device__ __forceinline__ void wait_async0() {
#if __has_builtin(__builtin_amdgcn_s_wait_asynccnt)
  __builtin_amdgcn_s_wait_asynccnt(0);
#else
  asm volatile("s_wait_asynccnt 0x0" ::: "memory");
#endif
}

// ---------------------------------------------------------------------------
// Fragment composer: two 16-byte LDS chunks -> one 16xbf16 WMMA operand
// ---------------------------------------------------------------------------
__device__ __forceinline__ v16bf frag_cat(const __bf16* p0, const __bf16* p1) {
  v8bf lo = *(const v8bf*)p0;
  v8bf hi = *(const v8bf*)p1;
  return __builtin_shufflevector(lo, hi, 0, 1, 2, 3, 4, 5, 6, 7,
                                         8, 9, 10, 11, 12, 13, 14, 15);
}

__device__ __forceinline__ float gelu_exact(float v) {
  return 0.5f * v * (1.0f + erff(v * 0.70710678118654752f));
}

// ---------------------------------------------------------------------------
// 1) Router: one wave (32 lanes) per token. logits = x@Wr + br, top-1 + gate.
// ---------------------------------------------------------------------------
__global__ void moe_router_kernel(const float* __restrict__ x,
                                  const float* __restrict__ Wr,
                                  const float* __restrict__ br,
                                  int* __restrict__ idx,
                                  float* __restrict__ gate,
                                  int* __restrict__ counts) {
  const int wave = threadIdx.x >> 5;
  const int lane = threadIdx.x & 31;
  const int tok  = blockIdx.x * 8 + wave;
  if (tok >= NN) return;

  const float* xr = x + (size_t)tok * DD;
  float acc[EE];
#pragma unroll
  for (int e = 0; e < EE; ++e) acc[e] = 0.0f;

  for (int d = lane; d < DD; d += 32) {
    const float xv = xr[d];
    const float4* w = (const float4*)(Wr + (size_t)d * EE);
    const float4 w0 = w[0];
    const float4 w1 = w[1];
    acc[0] += xv * w0.x; acc[1] += xv * w0.y;
    acc[2] += xv * w0.z; acc[3] += xv * w0.w;
    acc[4] += xv * w1.x; acc[5] += xv * w1.y;
    acc[6] += xv * w1.z; acc[7] += xv * w1.w;
  }
#pragma unroll
  for (int off = 16; off > 0; off >>= 1) {
#pragma unroll
    for (int e = 0; e < EE; ++e) acc[e] += __shfl_xor(acc[e], off, 32);
  }
  if (lane == 0) {
    float l[EE];
    float mx = -3.4e38f;
    int am = 0;
#pragma unroll
    for (int e = 0; e < EE; ++e) {
      l[e] = acc[e] + br[e];
      if (l[e] > mx) { mx = l[e]; am = e; }   // first-max tie-break like argmax
    }
    float s = 0.0f;
#pragma unroll
    for (int e = 0; e < EE; ++e) s += __expf(l[e] - mx);
    idx[tok]  = am;
    gate[tok] = 1.0f / s;                      // softmax prob of the max logit
    atomicAdd(&counts[am], 1);
  }
}

// ---------------------------------------------------------------------------
// 2) Exclusive scan over E=8 expert counts
// ---------------------------------------------------------------------------
__global__ void moe_scan_kernel(const int* __restrict__ counts,
                                int* __restrict__ offs,
                                int* __restrict__ cursor) {
  if (threadIdx.x == 0 && blockIdx.x == 0) {
    int run = 0;
    for (int e = 0; e < EE; ++e) {
      offs[e] = run;
      cursor[e] = run;
      run += counts[e];
    }
  }
}

// ---------------------------------------------------------------------------
// 3) Scatter token ids into per-expert buckets
// ---------------------------------------------------------------------------
__global__ void moe_scatter_kernel(const int* __restrict__ idx,
                                   int* __restrict__ cursor,
                                   int* __restrict__ bucket) {
  const int i = blockIdx.x * blockDim.x + threadIdx.x;
  if (i >= NN) return;
  const int e = idx[i];
  const int pos = atomicAdd(&cursor[e], 1);
  bucket[pos] = i;
}

// ---------------------------------------------------------------------------
// 4) GEMM1: H = gelu(Xg @ W1[e] + b1[e])  (bf16 WMMA, f32 accum, bf16 H)
//    Block tile 128x128, K over D in steps of 32, double-buffered LDS.
//    8 waves in 4(M) x 2(N); each wave owns 2x4 WMMA 16x16 tiles.
//    Tail rows are CLAMPED into the segment (no EXEC divergence in staging);
//    invalid rows are masked only at the H store.
// ---------------------------------------------------------------------------
__global__ __launch_bounds__(256)
void moe_ffn1_kernel(const float* __restrict__ x,
                     const float* __restrict__ W1,
                     const float* __restrict__ b1,
                     const int* __restrict__ bucket,
                     const int* __restrict__ offs,
                     const int* __restrict__ counts,
                     __bf16* __restrict__ H) {
  const int e    = blockIdx.z;
  const int cnt  = counts[e];
  const int mtb  = blockIdx.y * 128;
  if (mtb >= cnt) return;
  const int off  = offs[e];
  const int n0   = blockIdx.x * 128;

  __shared__ __bf16 As[2][128][40];   // [buf][token][k]
  __shared__ __bf16 Bs[2][128][40];   // [buf][F-col][k] (W1 transposed)
  __shared__ int    rowTok[128];

  const int t = threadIdx.x;
  if (t < 128) {
    const int m = mtb + t;
    rowTok[t] = (m < cnt) ? bucket[off + m] : -1;
  }
  __syncthreads();

  const float* W1e = W1 + (size_t)e * DD * FF;
  const float* b1e = b1 + (size_t)e * FF;

  // ---- hoisted staging pointers (rows clamped: always a valid token) ----
  const int rmax = cnt - mtb - 1;     // >= 0; may exceed 127
  const int ar = t >> 3;              // 0..31, A rows ar + p*32
  const int ac = (t & 7) << 2;        // A col group (4 floats)
  const float* aptr[4];
#pragma unroll
  for (int p = 0; p < 4; ++p) {
    int r = ar + p * 32;
    if (r > rmax) r = rmax;
    aptr[p] = x + (size_t)rowTok[r] * DD + ac;
  }
  const int bkp = t >> 5;             // 0..7, K-pair index base
  const int bc  = (t & 31) << 2;      // B col group (4 cols)

  auto stage = [&](int buf, int k0) {
#pragma unroll
    for (int p = 0; p < 4; ++p) {
      const float4 v = *(const float4*)(aptr[p] + k0);
      v4bf o = {(__bf16)v.x, (__bf16)v.y, (__bf16)v.z, (__bf16)v.w};
      *(v4bf*)&As[buf][ar + p * 32][ac] = o;
    }
#pragma unroll
    for (int p = 0; p < 2; ++p) {
      const int kp = bkp + p * 8;     // 0..15 -> K rows 2kp, 2kp+1
      const float* w0 = W1e + (size_t)(k0 + 2 * kp) * FF + n0 + bc;
      const float4 wa = *(const float4*)w0;
      const float4 wb = *(const float4*)(w0 + FF);
      v2bf q0 = {(__bf16)wa.x, (__bf16)wb.x};
      v2bf q1 = {(__bf16)wa.y, (__bf16)wb.y};
      v2bf q2 = {(__bf16)wa.z, (__bf16)wb.z};
      v2bf q3 = {(__bf16)wa.w, (__bf16)wb.w};
      *(v2bf*)&Bs[buf][bc + 0][2 * kp] = q0;
      *(v2bf*)&Bs[buf][bc + 1][2 * kp] = q1;
      *(v2bf*)&Bs[buf][bc + 2][2 * kp] = q2;
      *(v2bf*)&Bs[buf][bc + 3][2 * kp] = q3;
    }
  };

  const int lane = t & 31;
  const int wid  = t >> 5;
  const int wm   = wid & 3;
  const int wn   = wid >> 2;
  const int rsel = lane & 15;
  const int kbA  = (lane & 16) ? 8 : 0;
  const int kbB  = (lane & 16) ? 8 : 0;

  v8f acc[2][4] = {};

  stage(0, 0);
  __syncthreads();

  for (int k0 = 0, it = 0; k0 < DD; k0 += 32, ++it) {
    const int cur = it & 1;
    if (k0 + 32 < DD) stage(cur ^ 1, k0 + 32);   // overlap next stage w/ compute

    v16bf afrag[2];
#pragma unroll
    for (int i = 0; i < 2; ++i) {
      const int row = wm * 32 + i * 16 + rsel;
      afrag[i] = frag_cat(&As[cur][row][kbA], &As[cur][row][kbA + 16]);
    }
    v16bf bfrag[4];
#pragma unroll
    for (int j = 0; j < 4; ++j) {
      const int col = wn * 64 + j * 16 + rsel;
      bfrag[j] = frag_cat(&Bs[cur][col][kbB], &Bs[cur][col][kbB + 16]);
    }
#pragma unroll
    for (int i = 0; i < 2; ++i)
#pragma unroll
      for (int j = 0; j < 4; ++j)
        acc[i][j] = __builtin_amdgcn_wmma_f32_16x16x32_bf16(
            false, afrag[i], false, bfrag[j], (short)0, acc[i][j], false, false);
    __syncthreads();
  }

  // ---- epilogue: bias + exact gelu, write bf16 H at bucketed row ----
  const int segbase = off + mtb;
  float bias[4];
#pragma unroll
  for (int j = 0; j < 4; ++j)
    bias[j] = b1e[n0 + wn * 64 + j * 16 + rsel];

#pragma unroll
  for (int i = 0; i < 2; ++i) {
#pragma unroll
    for (int j = 0; j < 4; ++j) {
#pragma unroll
      for (int p = 0; p < 8; ++p) {
        const int row = wm * 32 + i * 16 + p + ((lane & 16) ? 8 : 0);
        const int col = wn * 64 + j * 16 + rsel;
        if (rowTok[row] >= 0) {
          float h = acc[i][j][p] + bias[j];
          H[(size_t)(segbase + row) * FF + n0 + col] = (__bf16)gelu_exact(h);
        }
      }
    }
  }
}

// ---------------------------------------------------------------------------
// 5) GEMM2: out[tok] = (H @ W2[e] + b2[e]) * gate[tok]  (scatter rows)
//    A operand (bf16 H) staged via GLOBAL_LOAD_ASYNC_TO_LDS_B128 (ASYNCcnt).
// ---------------------------------------------------------------------------
__global__ __launch_bounds__(256)
void moe_ffn2_kernel(const __bf16* __restrict__ H,
                     const float* __restrict__ W2,
                     const float* __restrict__ b2,
                     const int* __restrict__ bucket,
                     const int* __restrict__ offs,
                     const int* __restrict__ counts,
                     const float* __restrict__ gate,
                     float* __restrict__ out) {
  const int e    = blockIdx.z;
  const int cnt  = counts[e];
  const int mtb  = blockIdx.y * 128;
  if (mtb >= cnt) return;
  const int off  = offs[e];
  const int n0   = blockIdx.x * 128;

  __shared__ __bf16 As[2][128][40];
  __shared__ __bf16 Bs[2][128][40];
  __shared__ int    rowTok[128];
  __shared__ float  gateS[128];

  const int t = threadIdx.x;
  if (t < 128) {
    const int m = mtb + t;
    const int tok = (m < cnt) ? bucket[off + m] : -1;
    rowTok[t] = tok;
    gateS[t]  = (tok >= 0) ? gate[tok] : 0.0f;
  }
  __syncthreads();

  const float* W2e = W2 + (size_t)e * FF * DD;
  const float* b2e = b2 + (size_t)e * DD;

  // ---- hoisted staging pointers: clamp tail rows inside the expert segment
  //      so the async copy runs with full EXEC and never goes OOB.
  const int ar = t >> 2;              // 0..63, A rows ar + p*64
  const int ac = (t & 3) << 3;        // 8 bf16 (16B) per thread
  const __bf16* hptr[2];
#pragma unroll
  for (int p = 0; p < 2; ++p) {
    int m = mtb + ar + p * 64;
    if (m >= cnt) m = cnt - 1;
    hptr[p] = H + (size_t)(off + m) * FF + ac;
  }
  const int bkp = t >> 5;
  const int bc  = (t & 31) << 2;

  auto stage = [&](int buf, int k0) {
#pragma unroll
    for (int p = 0; p < 2; ++p)       // async DMA: 16B global -> LDS, no VGPRs
      async_copy_b128(hptr[p] + k0, &As[buf][ar + p * 64][ac]);
#pragma unroll
    for (int p = 0; p < 2; ++p) {
      const int kp = bkp + p * 8;
      const float* w0 = W2e + (size_t)(k0 + 2 * kp) * DD + n0 + bc;
      const float4 wa = *(const float4*)w0;
      const float4 wb = *(const float4*)(w0 + DD);
      v2bf q0 = {(__bf16)wa.x, (__bf16)wb.x};
      v2bf q1 = {(__bf16)wa.y, (__bf16)wb.y};
      v2bf q2 = {(__bf16)wa.z, (__bf16)wb.z};
      v2bf q3 = {(__bf16)wa.w, (__bf16)wb.w};
      *(v2bf*)&Bs[buf][bc + 0][2 * kp] = q0;
      *(v2bf*)&Bs[buf][bc + 1][2 * kp] = q1;
      *(v2bf*)&Bs[buf][bc + 2][2 * kp] = q2;
      *(v2bf*)&Bs[buf][bc + 3][2 * kp] = q3;
    }
  };

  const int lane = t & 31;
  const int wid  = t >> 5;
  const int wm   = wid & 3;
  const int wn   = wid >> 2;
  const int rsel = lane & 15;
  const int kbA  = (lane & 16) ? 8 : 0;
  const int kbB  = (lane & 16) ? 8 : 0;

  v8f acc[2][4] = {};

  stage(0, 0);
  wait_async0();
  __syncthreads();

  for (int k0 = 0, it = 0; k0 < FF; k0 += 32, ++it) {
    const int cur = it & 1;
    if (k0 + 32 < FF) stage(cur ^ 1, k0 + 32);

    v16bf afrag[2];
#pragma unroll
    for (int i = 0; i < 2; ++i) {
      const int row = wm * 32 + i * 16 + rsel;
      afrag[i] = frag_cat(&As[cur][row][kbA], &As[cur][row][kbA + 16]);
    }
    v16bf bfrag[4];
#pragma unroll
    for (int j = 0; j < 4; ++j) {
      const int col = wn * 64 + j * 16 + rsel;
      bfrag[j] = frag_cat(&Bs[cur][col][kbB], &Bs[cur][col][kbB + 16]);
    }
#pragma unroll
    for (int i = 0; i < 2; ++i)
#pragma unroll
      for (int j = 0; j < 4; ++j)
        acc[i][j] = __builtin_amdgcn_wmma_f32_16x16x32_bf16(
            false, afrag[i], false, bfrag[j], (short)0, acc[i][j], false, false);

    wait_async0();                    // next-buffer async writes landed
    __syncthreads();
  }

  // ---- epilogue: bias + gate scale, scatter fp32 rows to out ----
  float bias[4];
#pragma unroll
  for (int j = 0; j < 4; ++j)
    bias[j] = b2e[n0 + wn * 64 + j * 16 + rsel];

#pragma unroll
  for (int i = 0; i < 2; ++i) {
#pragma unroll
    for (int j = 0; j < 4; ++j) {
#pragma unroll
      for (int p = 0; p < 8; ++p) {
        const int row = wm * 32 + i * 16 + p + ((lane & 16) ? 8 : 0);
        const int col = wn * 64 + j * 16 + rsel;
        const int tok = rowTok[row];
        if (tok >= 0) {
          const float y = acc[i][j][p] + bias[j];
          out[(size_t)tok * DD + n0 + col] = y * gateS[row];
        }
      }
    }
  }
}

// ---------------------------------------------------------------------------
// Host-side orchestration
// ---------------------------------------------------------------------------
extern "C" void kernel_launch(void* const* d_in, const int* in_sizes, int n_in,
                              void* d_out, int out_size, void* d_ws, size_t ws_size,
                              hipStream_t stream) {
  const float* x  = (const float*)d_in[0];   // [B,S,D]
  const float* W1 = (const float*)d_in[1];   // [E,D,F]
  const float* b1 = (const float*)d_in[2];   // [E,F]
  const float* W2 = (const float*)d_in[3];   // [E,F,D]
  const float* b2 = (const float*)d_in[4];   // [E,D]
  const float* Wr = (const float*)d_in[5];   // [D,E]
  const float* br = (const float*)d_in[6];   // [E]
  float* out = (float*)d_out;                // [B,S,D]

  // Workspace carve-up
  char* w = (char*)d_ws;
  int*   idx    = (int*)w;    w += (size_t)NN * sizeof(int);
  float* gate   = (float*)w;  w += (size_t)NN * sizeof(float);
  int*   counts = (int*)w;    w += 16 * sizeof(int);
  int*   offs   = (int*)w;    w += 16 * sizeof(int);
  int*   cursor = (int*)w;    w += 16 * sizeof(int);
  int*   bucket = (int*)w;    w += (size_t)NN * sizeof(int);
  uintptr_t up = ((uintptr_t)w + 255) & ~(uintptr_t)255;
  __bf16* H = (__bf16*)up;                  // [N, F] bf16  (64 MB)

  hipMemsetAsync(counts, 0, 16 * sizeof(int), stream);

  moe_router_kernel<<<NN / 8, 256, 0, stream>>>(x, Wr, br, idx, gate, counts);
  moe_scan_kernel<<<1, 32, 0, stream>>>(counts, offs, cursor);
  moe_scatter_kernel<<<NN / 256, 256, 0, stream>>>(idx, cursor, bucket);

  dim3 g1(FF / 128, NN / 128, EE);   // (32, 64, 8), inactive tiles early-exit
  moe_ffn1_kernel<<<g1, 256, 0, stream>>>(x, W1, b1, bucket, offs, counts, H);

  dim3 g2(DD / 128, NN / 128, EE);   // (8, 64, 8)
  moe_ffn2_kernel<<<g2, 256, 0, stream>>>(H, W2, b2, bucket, offs, counts, gate, out);
}
